// self_attention_fc_4475355923112
// MI455X (gfx1250) — matrix operand order repaired and checked by reading, not compiled, not run on hardware
//
#include <hip/hip_runtime.h>
#include <hip/hip_bf16.h>

// ---------------------------------------------------------------------------
// Problem constants (from reference): B=8, C_IN=C_OUT=128, N=16384, fp32.
// ---------------------------------------------------------------------------
constexpr int BN = 8;      // batch
constexpr int CD = 128;    // channels (in == out)
constexpr int NL = 16384;  // sequence length
constexpr int NCOL = 32;   // columns per workgroup in the big GEMMs
constexpr int ROWP = 136;  // LDS row pitch in ushorts: 128 k + 8 pad (bank-spread)

typedef unsigned short u16;
typedef __attribute__((ext_vector_type(16))) __bf16  v16bf;
typedef __attribute__((ext_vector_type(8)))  float   v8f;
typedef __attribute__((ext_vector_type(8)))  u16     ushort8;

// WMMA operand: two 16-byte halves land directly in the operand VGPR quads.
union Frag { ushort8 h[2]; v16bf v; };

__device__ __forceinline__ u16 f32_to_bf16(float f) {
    union { float f; unsigned int u; } cv; cv.f = f;
    unsigned int u = cv.u;
    unsigned int r = u + 0x7FFFu + ((u >> 16) & 1u);   // round-to-nearest-even
    return (u16)(r >> 16);
}

// mish(h) = h * tanh(softplus(h)) = h * ((1+e^h)^2 - 1) / ((1+e^h)^2 + 1)
// One v_exp_f32 + one v_rcp_f32 per element.  Clamp keeps (1+e)^2 finite;
// for h > 20 the ratio is 1 to fp32 precision (mish -> h), for h -> -inf -> 0.
__device__ __forceinline__ float mish(float h) {
    float e = __expf(fminf(h, 20.0f));
    float t = 1.0f + e;
    t = t * t;                               // exp(2*softplus(h))
    return h * (t - 1.0f) * __builtin_amdgcn_rcpf(t + 1.0f);
}

// ---------------------------------------------------------------------------
// Kernel 1: BN affine (scale/shift) + W fp32 -> bf16
// ---------------------------------------------------------------------------
__global__ void prep_kernel(const float* __restrict__ W,
                            const float* __restrict__ gamma,
                            const float* __restrict__ bn_beta,
                            const float* __restrict__ rmean,
                            const float* __restrict__ rvar,
                            float* __restrict__ scale,
                            float* __restrict__ shift,
                            u16* __restrict__ Wbf) {
    int t = threadIdx.x;
    if (t < CD) {
        float s = gamma[t] * rsqrtf(rvar[t] + 1e-5f);
        scale[t] = s;
        shift[t] = bn_beta[t] - rmean[t] * s;
    }
    for (int i = t; i < CD * CD; i += blockDim.x)
        Wbf[i] = f32_to_bf16(W[i]);
}

__global__ void zero_energy_kernel(float* __restrict__ energy) {
    int i = blockIdx.x * blockDim.x + threadIdx.x;
    if (i < BN * CD * CD) energy[i] = 0.0f;
}

// ---------------------------------------------------------------------------
// Kernel 2: h = W @ X + BN affine + Mish, writing fp32 AND bf16 copies.
// 256 threads = 8 waves; wave m owns out-channel tile m; block covers a
// 32-column strip with 2 accumulators/wave (A fragment reused).
// Whole K=128 x 32-col X panel staged once into [n][k] bf16 LDS (padded rows)
// so each B fragment is two contiguous ds_load_b128 per lane.
// ---------------------------------------------------------------------------
__global__ __launch_bounds__(256)
void conv_bn_mish_kernel(const float* __restrict__ x, const float* __restrict__ y,
                         const u16* __restrict__ Wbf,
                         const float* __restrict__ scale, const float* __restrict__ shift,
                         float* __restrict__ qx, float* __restrict__ ky,
                         u16* __restrict__ qxbf, u16* __restrict__ kybf) {
    const float* X    = blockIdx.z ? y : x;
    float*       Qf   = blockIdx.z ? ky : qx;
    u16*         Qb16 = blockIdx.z ? kybf : qxbf;
    const int b    = blockIdx.y;
    const int col0 = blockIdx.x * NCOL;
    const int wave = threadIdx.x >> 5;
    const int lane = threadIdx.x & 31;

    __shared__ u16 tileT[NCOL * ROWP];           // [n][k] bf16, padded

    const float* Xb = X + (size_t)b * CD * NL;

    // stage: 128 k x 32 n, packed k-pairs -> one b32 LDS store each
    {
        const int n = threadIdx.x & 31;
        unsigned int* row = reinterpret_cast<unsigned int*>(tileT + n * ROWP);
        #pragma unroll
        for (int it = 0; it < 8; ++it) {
            int p = (threadIdx.x >> 5) + 8 * it;             // k-pair 0..63
            float f0 = Xb[(size_t)(2 * p)     * NL + col0 + n];
            float f1 = Xb[(size_t)(2 * p + 1) * NL + col0 + n];
            row[p] = (unsigned int)f32_to_bf16(f0) | ((unsigned int)f32_to_bf16(f1) << 16);
        }
    }
    __syncthreads();

    const int arow = wave * 16 + (lane & 15);
    const int kb8  = (lane >> 4) * 8;
    const int ncol = lane & 15;
    const int kb16 = (lane >> 4) * 16;

    v8f acc0 = {0.f, 0.f, 0.f, 0.f, 0.f, 0.f, 0.f, 0.f};
    v8f acc1 = acc0;

    #pragma unroll
    for (int k0 = 0; k0 < CD; k0 += 32) {
        Frag a, b0, b1;
        a.h[0] = *reinterpret_cast<const ushort8*>(Wbf + arow * CD + k0 + kb16);
        a.h[1] = *reinterpret_cast<const ushort8*>(Wbf + arow * CD + k0 + kb16 + 8);
        const u16* r0 = tileT + ncol * ROWP        + k0 + kb16;
        const u16* r1 = tileT + (ncol + 16) * ROWP + k0 + kb16;
        b0.h[0] = *reinterpret_cast<const ushort8*>(r0);
        b0.h[1] = *reinterpret_cast<const ushort8*>(r0 + 8);
        b1.h[0] = *reinterpret_cast<const ushort8*>(r1);
        b1.h[1] = *reinterpret_cast<const ushort8*>(r1 + 8);
        acc0 = __builtin_amdgcn_wmma_f32_16x16x32_bf16(false, a.v, false, b0.v,
                                                       (short)0, acc0, false, false);
        acc1 = __builtin_amdgcn_wmma_f32_16x16x32_bf16(false, a.v, false, b1.v,
                                                       (short)0, acc1, false, false);
    }

    const int rbase = wave * 16 + (lane >> 4) * 8;
    #pragma unroll
    for (int t = 0; t < 2; ++t) {
        v8f acc = t ? acc1 : acc0;
        int nout = col0 + t * 16 + (lane & 15);
        #pragma unroll
        for (int r = 0; r < 8; ++r) {
            int row = rbase + r;
            float h = acc[r] * scale[row] + shift[row];
            float m = mish(h);
            size_t idx = ((size_t)b * CD + row) * NL + nout;
            Qf[idx]   = m;
            Qb16[idx] = f32_to_bf16(m);
        }
    }
}

// ---------------------------------------------------------------------------
// Kernel 3: energy[b,d,e] = sum_n qx[b,d,n]*ky[b,e,n].  Reads the bf16 copies
// directly: per iteration 4x global_load_b128 + 1 WMMA, zero conversion VALU.
// split-K=8 with f32 global atomics (512 blocks for occupancy).
// ---------------------------------------------------------------------------
__global__ __launch_bounds__(256)
void energy_kernel(const u16* __restrict__ qxbf, const u16* __restrict__ kybf,
                   float* __restrict__ energy) {
    const int b      = blockIdx.z;
    const int kchunk = blockIdx.y;
    const int dtile  = blockIdx.x;
    const int etile  = threadIdx.x >> 5;
    const int lane   = threadIdx.x & 31;

    const u16* Qb = qxbf + (size_t)b * CD * NL;
    const u16* Kb = kybf + (size_t)b * CD * NL;

    const int arow = dtile * 16 + (lane & 15);
    const int erow = etile * 16 + (lane & 15);
    const int kb8  = (lane >> 4) * 8;
    const int kb16 = (lane >> 4) * 16;

    const u16* ap = Qb + (size_t)arow * NL + kb8;
    const u16* bp = Kb + (size_t)erow * NL + kb8;

    v8f acc = {0.f, 0.f, 0.f, 0.f, 0.f, 0.f, 0.f, 0.f};

    const int kbeg = kchunk * (NL / 8), kend = kbeg + (NL / 8);
    #pragma unroll 2
    for (int k0 = kbeg; k0 < kend; k0 += 32) {
        __builtin_prefetch(ap + k0 + 128, 0, 1);        // global_prefetch ahead
        __builtin_prefetch(bp + k0 + 128, 0, 1);
        Frag a, bf;
        a.h[0]  = *reinterpret_cast<const ushort8*>(ap + k0);
        a.h[1]  = *reinterpret_cast<const ushort8*>(ap + k0 + 16);
        bf.h[0] = *reinterpret_cast<const ushort8*>(bp + k0);
        bf.h[1] = *reinterpret_cast<const ushort8*>(bp + k0 + 16);
        acc = __builtin_amdgcn_wmma_f32_16x16x32_bf16(false, a.v, false, bf.v,
                                                      (short)0, acc, false, false);
    }

    const int ecol  = etile * 16 + (lane & 15);
    const int dbase = dtile * 16 + (lane >> 4) * 8;
    #pragma unroll
    for (int r = 0; r < 8; ++r)
        atomicAdd(&energy[((size_t)b * CD + dbase + r) * CD + ecol], acc[r]);
}

// ---------------------------------------------------------------------------
// Kernel 4: dual softmax -> bf16 attention matrices.
// ---------------------------------------------------------------------------
__device__ __forceinline__ float block_max(float v, float* red) {
    int t = threadIdx.x;
    red[t] = v; __syncthreads();
    for (int s = 64; s > 0; s >>= 1) {
        if (t < s) red[t] = fmaxf(red[t], red[t + s]);
        __syncthreads();
    }
    float r = red[0]; __syncthreads();
    return r;
}
__device__ __forceinline__ float block_sum(float v, float* red) {
    int t = threadIdx.x;
    red[t] = v; __syncthreads();
    for (int s = 64; s > 0; s >>= 1) {
        if (t < s) red[t] += red[t + s];
        __syncthreads();
    }
    float r = red[0]; __syncthreads();
    return r;
}

__global__ __launch_bounds__(128)
void softmax_kernel(const float* __restrict__ energy,
                    u16* __restrict__ att_xy, u16* __restrict__ att_yx) {
    const int b = blockIdx.y, d = blockIdx.x, e = threadIdx.x;
    __shared__ float red[128];
    const float* E = energy + (size_t)b * CD * CD;

    float rv = E[d * CD + e];   // row d    -> att_xy
    float cv = E[e * CD + d];   // col d    -> att_yx

    float rm = block_max(rv, red);
    float re = __expf(rv - rm);                 // v_exp_f32
    float rs = block_sum(re, red);
    att_xy[((size_t)b * CD + d) * CD + e] =
        f32_to_bf16(re * __builtin_amdgcn_rcpf(rs));

    float cm = block_max(cv, red);
    float ce = __expf(cv - cm);
    float cs = block_sum(ce, red);
    att_yx[((size_t)b * CD + d) * CD + e] =
        f32_to_bf16(ce * __builtin_amdgcn_rcpf(cs));
}

// ---------------------------------------------------------------------------
// Kernel 5: out = beta * (att @ Q) + Q.  B panel staged from the bf16 copy
// (pure repack, no conversion); fused beta-residual epilogue from fp32 Q.
// ---------------------------------------------------------------------------
__global__ __launch_bounds__(256)
void out_kernel(const u16* __restrict__ att_xy, const u16* __restrict__ att_yx,
                const float* __restrict__ qx, const float* __restrict__ ky,
                const u16* __restrict__ qxbf, const u16* __restrict__ kybf,
                const float* __restrict__ beta_p, float* __restrict__ out) {
    const u16*   Aall = blockIdx.z ? att_yx : att_xy;
    const float* Qf   = blockIdx.z ? ky : qx;
    const u16*   Qh   = blockIdx.z ? kybf : qxbf;
    float*       O    = out + (size_t)blockIdx.z * BN * CD * NL;
    const float beta = *beta_p;

    const int b    = blockIdx.y;
    const int col0 = blockIdx.x * NCOL;
    const int wave = threadIdx.x >> 5;
    const int lane = threadIdx.x & 31;

    __shared__ u16 tileT[NCOL * ROWP];

    const u16*   Ab  = Aall + (size_t)b * CD * CD;
    const u16*   Qhb = Qh   + (size_t)b * CD * NL;
    const float* Qfb = Qf   + (size_t)b * CD * NL;

    // stage: transpose-repack bf16 panel, one b32 LDS store per element pair
    {
        const int n = threadIdx.x & 31;
        unsigned int* row = reinterpret_cast<unsigned int*>(tileT + n * ROWP);
        #pragma unroll
        for (int it = 0; it < 8; ++it) {
            int p = (threadIdx.x >> 5) + 8 * it;
            unsigned int lo = Qhb[(size_t)(2 * p)     * NL + col0 + n];
            unsigned int hi = Qhb[(size_t)(2 * p + 1) * NL + col0 + n];
            row[p] = lo | (hi << 16);
        }
    }
    __syncthreads();

    const int arow = wave * 16 + (lane & 15);
    const int kb8  = (lane >> 4) * 8;
    const int ncol = lane & 15;
    const int kb16 = (lane >> 4) * 16;

    v8f acc0 = {0.f, 0.f, 0.f, 0.f, 0.f, 0.f, 0.f, 0.f};
    v8f acc1 = acc0;

    #pragma unroll
    for (int k0 = 0; k0 < CD; k0 += 32) {
        Frag a, b0, b1;
        a.h[0] = *reinterpret_cast<const ushort8*>(Ab + arow * CD + k0 + kb16);
        a.h[1] = *reinterpret_cast<const ushort8*>(Ab + arow * CD + k0 + kb16 + 8);
        const u16* r0 = tileT + ncol * ROWP        + k0 + kb16;
        const u16* r1 = tileT + (ncol + 16) * ROWP + k0 + kb16;
        b0.h[0] = *reinterpret_cast<const ushort8*>(r0);
        b0.h[1] = *reinterpret_cast<const ushort8*>(r0 + 8);
        b1.h[0] = *reinterpret_cast<const ushort8*>(r1);
        b1.h[1] = *reinterpret_cast<const ushort8*>(r1 + 8);
        acc0 = __builtin_amdgcn_wmma_f32_16x16x32_bf16(false, a.v, false, b0.v,
                                                       (short)0, acc0, false, false);
        acc1 = __builtin_amdgcn_wmma_f32_16x16x32_bf16(false, a.v, false, b1.v,
                                                       (short)0, acc1, false, false);
    }

    const int rbase = wave * 16 + (lane >> 4) * 8;
    #pragma unroll
    for (int t = 0; t < 2; ++t) {
        v8f acc = t ? acc1 : acc0;
        int nout = col0 + t * 16 + (lane & 15);
        #pragma unroll
        for (int r = 0; r < 8; ++r) {
            int row = rbase + r;
            size_t idx = ((size_t)b * CD + row) * NL + nout;
            O[idx] = beta * acc[r] + Qfb[(size_t)row * NL + nout];
        }
    }
}

// ---------------------------------------------------------------------------
// Host launcher
// ---------------------------------------------------------------------------
extern "C" void kernel_launch(void* const* d_in, const int* in_sizes, int n_in,
                              void* d_out, int out_size, void* d_ws, size_t ws_size,
                              hipStream_t stream) {
    const float* x       = (const float*)d_in[0];
    const float* y       = (const float*)d_in[1];
    const float* W       = (const float*)d_in[2];
    const float* gamma   = (const float*)d_in[3];
    const float* bn_beta = (const float*)d_in[4];
    const float* rmean   = (const float*)d_in[5];
    const float* rvar    = (const float*)d_in[6];
    const float* beta    = (const float*)d_in[7];
    float* out = (float*)d_out;

    // workspace layout
    char* ws = (char*)d_ws;
    float* scale  = (float*)ws;                              // 128 f32
    float* shift  = scale + CD;                              // 128 f32
    u16*   Wbf    = (u16*)(shift + CD);                      // 128*128 bf16
    float* qx     = (float*)(ws + 33792);                    // 8*128*16384 f32
    float* ky     = qx + (size_t)BN * CD * NL;               // 8*128*16384 f32
    u16*   qxbf   = (u16*)(ky + (size_t)BN * CD * NL);       // bf16 copy
    u16*   kybf   = qxbf + (size_t)BN * CD * NL;             // bf16 copy
    float* energy = (float*)(kybf + (size_t)BN * CD * NL);   // 8*128*128 f32
    u16*   att_xy = (u16*)(energy + (size_t)BN * CD * CD);
    u16*   att_yx = att_xy + (size_t)BN * CD * CD;

    prep_kernel<<<1, 256, 0, stream>>>(W, gamma, bn_beta, rmean, rvar, scale, shift, Wbf);
    zero_energy_kernel<<<(BN * CD * CD + 255) / 256, 256, 0, stream>>>(energy);
    conv_bn_mish_kernel<<<dim3(NL / NCOL, BN, 2), 256, 0, stream>>>(
        x, y, Wbf, scale, shift, qx, ky, qxbf, kybf);
    energy_kernel<<<dim3(8, 8, BN), 256, 0, stream>>>(qxbf, kybf, energy);
    softmax_kernel<<<dim3(CD, BN), 128, 0, stream>>>(energy, att_xy, att_yx);
    out_kernel<<<dim3(NL / NCOL, BN, 2), 256, 0, stream>>>(
        att_xy, att_yx, qx, ky, qxbf, kybf, beta, out);
}
